// MoE_57775900066388
// MI455X (gfx1250) — matrix operand order repaired and checked by reading, not compiled, not run on hardware
//
#include <hip/hip_runtime.h>
#include <hip/hip_bf16.h>

#define T_TOK 4096   // B*S
#define DM    512    // d_model
#define DFF   2048   // d_ff
#define NE    8      // experts

typedef __attribute__((ext_vector_type(16))) __bf16       v16bf;
typedef __attribute__((ext_vector_type(8)))  float        v8f;
typedef __attribute__((ext_vector_type(4)))  unsigned int v4u;

union BFrag { v4u u[2]; v16bf b; };   // 32 bytes = 16 bf16

__device__ __forceinline__ unsigned short f2bf(float f) {
    unsigned int u = __builtin_bit_cast(unsigned int, f);
    u += 0x7fffu + ((u >> 16) & 1u);        // round-to-nearest-even
    return (unsigned short)(u >> 16);
}

// ---------------- kernel 0: zero the output (we accumulate with atomics) ---
__global__ void moe_zero_f32(float* __restrict__ p, int n) {
    int i = blockIdx.x * blockDim.x + threadIdx.x;
    if (i < n) p[i] = 0.f;
}

// ---------------- kernel 1: gating (one wave32 per token) ------------------
__global__ __launch_bounds__(256)
void moe_gate(const float* __restrict__ x, const float* __restrict__ Wg,
              const float* __restrict__ bg, float* __restrict__ combine) {
    const int wave = (blockIdx.x * blockDim.x + threadIdx.x) >> 5;
    const int lane = threadIdx.x & 31;
    if (wave >= T_TOK) return;

    float acc[NE];
#pragma unroll
    for (int e = 0; e < NE; ++e) acc[e] = 0.f;

    const float* xrow = x + (size_t)wave * DM;
    for (int d = lane; d < DM; d += 32) {
        float xv = xrow[d];
        const float* wrow = Wg + (size_t)d * NE;
#pragma unroll
        for (int e = 0; e < NE; ++e) acc[e] += xv * wrow[e];
    }
#pragma unroll
    for (int off = 16; off > 0; off >>= 1)
#pragma unroll
        for (int e = 0; e < NE; ++e) acc[e] += __shfl_xor(acc[e], off, 32);

    if (lane == 0) {
        float logit[NE], p[NE];
        float mx = -3.4e38f;
#pragma unroll
        for (int e = 0; e < NE; ++e) { logit[e] = acc[e] + bg[e]; mx = fmaxf(mx, logit[e]); }
        float s = 0.f;
#pragma unroll
        for (int e = 0; e < NE; ++e) { p[e] = __expf(logit[e] - mx); s += p[e]; }
#pragma unroll
        for (int e = 0; e < NE; ++e) p[e] /= s;
        // top-2, ties -> lower index first (matches lax.top_k)
        int i1 = 0;
#pragma unroll
        for (int e = 1; e < NE; ++e) if (p[e] > p[i1]) i1 = e;
        int i2 = -1;
#pragma unroll
        for (int e = 0; e < NE; ++e) if (e != i1 && (i2 < 0 || p[e] > p[i2])) i2 = e;
        float denom = p[i1] + p[i2];
        float* crow = combine + (size_t)wave * NE;
#pragma unroll
        for (int e = 0; e < NE; ++e)
            crow[e] = (e == i1) ? p[i1] / denom : ((e == i2) ? p[i2] / denom : 0.f);
    }
}

// ---------------- kernels 2-4: fp32 -> bf16 (with W transposes) ------------
__global__ void moe_cvt_x(const float* __restrict__ in, unsigned short* __restrict__ out, int n) {
    int i = blockIdx.x * blockDim.x + threadIdx.x;
    if (i < n) out[i] = f2bf(in[i]);
}
// W1 [E][d][F] fp32  ->  W1t [E][F][d] bf16   (columns of W1 become rows)
__global__ void moe_cvt_w1t(const float* __restrict__ in, unsigned short* __restrict__ out) {
    int i = blockIdx.x * blockDim.x + threadIdx.x;
    if (i >= NE * DM * DFF) return;
    int e = i / (DM * DFF), rem = i % (DM * DFF);
    int f = rem / DM, d = rem % DM;
    out[i] = f2bf(in[(size_t)e * DM * DFF + (size_t)d * DFF + f]);
}
// W2 [E][F][d] fp32  ->  W2t [E][d][F] bf16
__global__ void moe_cvt_w2t(const float* __restrict__ in, unsigned short* __restrict__ out) {
    int i = blockIdx.x * blockDim.x + threadIdx.x;
    if (i >= NE * DM * DFF) return;
    int e = i / (DM * DFF), rem = i % (DM * DFF);
    int d = rem / DFF, f = rem % DFF;
    out[i] = f2bf(in[(size_t)e * DFF * DM + (size_t)f * DM + d]);
}

// ---------------- kernel 5: fused expert FFN (bf16 WMMA, top-k sparse) -----
// grid = (T/16 token tiles, E experts); block = 256 = 8 waves.
__global__ __launch_bounds__(256)
void moe_ffn(const unsigned short* __restrict__ xb,
             const unsigned short* __restrict__ W1t,
             const unsigned short* __restrict__ W2t,
             const float* __restrict__ b1, const float* __restrict__ b2,
             const float* __restrict__ combine, float* __restrict__ out) {
    __shared__ __align__(16) unsigned short xs[16 * DM];   // 16 KB: x tile bf16
    __shared__ __align__(16) unsigned short hs[16 * 128];  // 4 KB: h chunk bf16
    __shared__ float sgate[16];

    const int tid   = threadIdx.x;
    const int lane  = tid & 31;
    const int w     = tid >> 5;        // wave 0..7
    const int lhalf = lane >> 4;       // 0 / 1
    const int lcol  = lane & 15;
    const int tile  = blockIdx.x;      // token tile
    const int e     = blockIdx.y;      // expert
    const int t0    = tile * 16;

    // cooperative x-tile load (16x512 bf16 = 1024 x 16B)
    {
        const v4u* src = (const v4u*)(xb + (size_t)t0 * DM);
        v4u* dst = (v4u*)xs;
#pragma unroll
        for (int i = tid; i < 1024; i += 256) dst[i] = src[i];
    }
    if (tid < 16) sgate[tid] = combine[(size_t)(t0 + tid) * NE + e];
    __syncthreads();

    // skip blocks where this expert is selected by none of the 16 tokens
    bool any = false;
#pragma unroll
    for (int m = 0; m < 16; ++m) any = any || (sgate[m] != 0.f);
    if (!any) return;

    const unsigned short* W1e = W1t + (size_t)e * DFF * DM;
    const unsigned short* W2e = W2t + (size_t)e * DM * DFF;

    v8f yacc[4];
#pragma unroll
    for (int t = 0; t < 4; ++t)
#pragma unroll
        for (int j = 0; j < 8; ++j) yacc[t][j] = 0.f;

    for (int c = 0; c < 16; ++c) {                 // F in chunks of 128
        // ---- GEMM1: h(16x16) = x(16x512) @ W1[:, c*128 + w*16 + lcol] ----
        const int fcol = c * 128 + w * 16 + lcol;  // this lane's B column
        const unsigned short* bp1 = W1e + (size_t)fcol * DM + lhalf * 8;
        if (c + 1 < 16) __builtin_prefetch(bp1 + (size_t)128 * DM, 0, 1);

        v8f hacc;
#pragma unroll
        for (int j = 0; j < 8; ++j) hacc[j] = 0.f;
#pragma unroll
        for (int kk = 0; kk < 16; ++kk) {          // K = 512 in steps of 32
            BFrag A, B;
            const unsigned short* ap = xs + lcol * DM + kk * 32 + lhalf * 8;
            A.u[0] = *(const v4u*)ap;              // K = khalf*8 + 0..7
            A.u[1] = *(const v4u*)(ap + 16);       // K = 16 + khalf*8 + 0..7
            B.u[0] = *(const v4u*)(bp1 + kk * 32); // K = kbase + 0..7
            B.u[1] = *(const v4u*)(bp1 + kk * 32 + 16);
            hacc = __builtin_amdgcn_wmma_f32_16x16x32_bf16(
                false, A.b, false, B.b, (short)0, hacc, false, false);
        }

        // bias + ReLU + bf16 -> LDS (C layout: M = r + 8*lhalf, N = lcol)
        const float b1v = b1[(size_t)e * DFF + c * 128 + w * 16 + lcol];
        __syncthreads();                           // prior chunk's h reads done
#pragma unroll
        for (int r = 0; r < 8; ++r) {
            int row = r + 8 * lhalf;
            float hv = hacc[r] + b1v;
            hv = hv > 0.f ? hv : 0.f;
            hs[row * 128 + w * 16 + lcol] = f2bf(hv);
        }
        __syncthreads();

        // ---- GEMM2: y(16x512) += h(16x128) @ W2[c*128.., :] --------------
#pragma unroll
        for (int kk = 0; kk < 4; ++kk) {           // K = 128 in steps of 32
            BFrag A;
            const unsigned short* ap = hs + lcol * 128 + kk * 32 + lhalf * 8;
            A.u[0] = *(const v4u*)ap;
            A.u[1] = *(const v4u*)(ap + 16);
#pragma unroll
            for (int t = 0; t < 4; ++t) {          // wave owns cols w*64..+63
                const int n = w * 64 + t * 16 + lcol;
                const unsigned short* bp2 =
                    W2e + (size_t)n * DFF + c * 128 + kk * 32 + lhalf * 8;
                BFrag B;
                B.u[0] = *(const v4u*)bp2;
                B.u[1] = *(const v4u*)(bp2 + 16);
                yacc[t] = __builtin_amdgcn_wmma_f32_16x16x32_bf16(
                    false, A.b, false, B.b, (short)0, yacc[t], false, false);
            }
        }
    }

    // epilogue: +b2, *gate, scatter-add (top-2 experts hit the same token)
#pragma unroll
    for (int t = 0; t < 4; ++t) {
        const int n = w * 64 + t * 16 + lcol;
        const float bias = b2[(size_t)e * DM + n];
#pragma unroll
        for (int r = 0; r < 8; ++r) {
            int m = r + 8 * lhalf;
            float g = sgate[m];
            if (g != 0.f)
                unsafeAtomicAdd(&out[(size_t)(t0 + m) * DM + n], g * (yacc[t][r] + bias));
        }
    }
}

// ---------------------------------------------------------------------------
extern "C" void kernel_launch(void* const* d_in, const int* in_sizes, int n_in,
                              void* d_out, int out_size, void* d_ws, size_t ws_size,
                              hipStream_t stream) {
    const float* x  = (const float*)d_in[0];
    const float* Wg = (const float*)d_in[1];
    const float* bg = (const float*)d_in[2];
    const float* W1 = (const float*)d_in[3];
    const float* b1 = (const float*)d_in[4];
    const float* W2 = (const float*)d_in[5];
    const float* b2 = (const float*)d_in[6];
    float* out = (float*)d_out;

    // workspace layout (bytes): combine | x bf16 | W1t bf16 | W2t bf16
    char* ws = (char*)d_ws;
    float*          combine = (float*)ws;                                   // 128 KB
    unsigned short* xb      = (unsigned short*)(ws + (size_t)131072);       // 4 MB
    unsigned short* W1t     = (unsigned short*)(ws + (size_t)131072 + 4194304);
    unsigned short* W2t     = (unsigned short*)(ws + (size_t)131072 + 4194304 + 16777216);

    moe_zero_f32<<<(T_TOK * DM + 255) / 256, 256, 0, stream>>>(out, T_TOK * DM);
    moe_gate<<<T_TOK / 8, 256, 0, stream>>>(x, Wg, bg, combine);
    moe_cvt_x<<<(T_TOK * DM + 255) / 256, 256, 0, stream>>>(x, xb, T_TOK * DM);
    moe_cvt_w1t<<<(NE * DM * DFF + 255) / 256, 256, 0, stream>>>(W1, W1t);
    moe_cvt_w2t<<<(NE * DM * DFF + 255) / 256, 256, 0, stream>>>(W2, W2t);
    moe_ffn<<<dim3(T_TOK / 16, NE), 256, 0, stream>>>(xb, W1t, W2t, b1, b2, combine, out);
}
